// MultipleSA_Head_53145925320763
// MI455X (gfx1250) — compile-verified, matrix operand order checked by reading, not run on hardware
//
#include <hip/hip_runtime.h>
#include <hip/hip_bf16.h>
#include <math.h>
#include <stdint.h>

// Problem constants (from reference): B=8, T=1024, E=1024, H=16, DH=64
#define BB  8
#define TT  1024
#define EE  1024
#define HH  16
#define DHH 64

typedef __attribute__((ext_vector_type(16))) __bf16 v16bf;
typedef __attribute__((ext_vector_type(8)))  __bf16 v8bf;
typedef __attribute__((ext_vector_type(8)))  float  v8f;

// ---------------------------------------------------------------------------
// D = A(16x32 bf16) x B(32x16 bf16) + C(16x16 f32)  -> v_wmma_f32_16x16x32_bf16
// ---------------------------------------------------------------------------
__device__ __forceinline__ v8f wmma_bf16(v16bf a, v16bf b, v8f c) {
  return __builtin_amdgcn_wmma_f32_16x16x32_bf16(false, a, false, b, (short)0, c,
                                                 false, false);
}

// ---------------------------------------------------------------------------
// Load a 16x32 (rows x K) bf16 fragment from a row-major matrix.
// ISA 16-bit fragment layout (wave32):
//   lanes 0-15 : row = row0+lane,    K chunks [k0+0..7]  and [k0+16..23]
//   lanes 16-31: row = row0+lane-16, K chunks [k0+8..15] and [k0+24..31]
// ---------------------------------------------------------------------------
__device__ __forceinline__ v16bf load_frag_g(const __bf16* __restrict__ base,
                                             int ld, int row0, int k0) {
  const int lane = threadIdx.x & 31;
  const int r    = lane & 15;
  const int kk   = k0 + ((lane >> 4) << 3);
  const __bf16* p = base + (size_t)(row0 + r) * (size_t)ld + kk;
  v8bf lo = *reinterpret_cast<const v8bf*>(p);
  v8bf hi = *reinterpret_cast<const v8bf*>(p + 16);
  v16bf out;
#pragma unroll
  for (int i = 0; i < 8; ++i) { out[i] = lo[i]; out[i + 8] = hi[i]; }
  return out;
}

// Same fragment pattern out of an LDS tile with row stride `ld` elements.
__device__ __forceinline__ v16bf load_frag_lds(const __bf16* base, int ld,
                                               int row0) {
  const int lane = threadIdx.x & 31;
  const int r    = lane & 15;
  const int kk   = (lane >> 4) << 3;
  const __bf16* p = base + (row0 + r) * ld + kk;
  v8bf lo = *reinterpret_cast<const v8bf*>(p);
  v8bf hi = *reinterpret_cast<const v8bf*>(p + 16);
  v16bf out;
#pragma unroll
  for (int i = 0; i < 8; ++i) { out[i] = lo[i]; out[i + 8] = hi[i]; }
  return out;
}

// 16B global -> LDS async copy (GLOBAL_LOAD_ASYNC_TO_LDS_B128, ASYNCcnt).
__device__ __forceinline__ void async_copy16(const __bf16* g, __bf16* l) {
  asm volatile("global_load_async_to_lds_b128 %0, %1, off"
               :: "v"((uint32_t)(uintptr_t)l), "v"((uint64_t)(uintptr_t)g)
               : "memory");
}

// ---------------------------------------------------------------------------
// fp32 -> bf16 bulk convert
// ---------------------------------------------------------------------------
__global__ void cvt_f32_to_bf16(const float* __restrict__ in,
                                __bf16* __restrict__ out, long n) {
  long i      = (long)blockIdx.x * blockDim.x + threadIdx.x;
  long stride = (long)gridDim.x * blockDim.x;
  for (; i < n; i += stride) out[i] = (__bf16)in[i];
}

// W[h][e][d] (fp32) -> Wt[h][d][e] (bf16): B^T rows become contiguous
__global__ void cvt_w_transpose(const float* __restrict__ W,
                                __bf16* __restrict__ Wt, long n) {
  long i      = (long)blockIdx.x * blockDim.x + threadIdx.x;
  long stride = (long)gridDim.x * blockDim.x;
  for (; i < n; i += stride) {
    long h   = i / ((long)EE * DHH);
    long rem = i - h * (long)EE * DHH;
    long e   = rem / DHH;
    long d   = rem - e * DHH;
    Wt[h * (long)DHH * EE + d * EE + e] = (__bf16)W[i];
  }
}

// ---------------------------------------------------------------------------
// Fused QKV projection: one wave per (b, h, 16-row tile of T) computes the
// 16x64 q, k and v tiles together: the x fragment is loaded ONCE per K-step
// and reused for 12 WMMAs (3 matrices x 4 DH tiles).
// V is stored transposed [DH][T] for the later P*V GEMM.
// ---------------------------------------------------------------------------
__global__ __launch_bounds__(32) void qkv_gemm_fused(
    const __bf16* __restrict__ xb,
    const __bf16* __restrict__ wqt, const __bf16* __restrict__ wkt,
    const __bf16* __restrict__ wvt,
    __bf16* __restrict__ qo, __bf16* __restrict__ ko, __bf16* __restrict__ vt) {
  const int tm   = blockIdx.x;           // 0..63 (T/16)
  const int h    = blockIdx.y;           // 0..15
  const int b    = blockIdx.z;
  const int lane = threadIdx.x & 31;

  const __bf16* A  = xb  + (size_t)b * TT * EE;
  const __bf16* Bq = wqt + (size_t)h * DHH * EE;
  const __bf16* Bk = wkt + (size_t)h * DHH * EE;
  const __bf16* Bv = wvt + (size_t)h * DHH * EE;

  v8f cq[4] = {}, ck[4] = {}, cv[4] = {};
  for (int k0 = 0; k0 < EE; k0 += 32) {
    v16bf a = load_frag_g(A, EE, tm * 16, k0);   // shared by all 12 WMMAs
#pragma unroll
    for (int j = 0; j < 4; ++j) {
      cq[j] = wmma_bf16(a, load_frag_g(Bq, EE, j * 16, k0), cq[j]);
      ck[j] = wmma_bf16(a, load_frag_g(Bk, EE, j * 16, k0), ck[j]);
      cv[j] = wmma_bf16(a, load_frag_g(Bv, EE, j * 16, k0), cv[j]);
    }
  }

  const int n  = lane & 15;
  const int ro = (lane >> 4) << 3;
  const size_t bh = (size_t)(b * HH + h);
#pragma unroll
  for (int j = 0; j < 4; ++j) {
#pragma unroll
    for (int r = 0; r < 8; ++r) {
      const int t  = tm * 16 + ro + r;
      const int dg = j * 16 + n;
      qo[(bh * TT + t) * DHH + dg] = (__bf16)cq[j][r];
      ko[(bh * TT + t) * DHH + dg] = (__bf16)ck[j][r];
      vt[(bh * DHH + dg) * TT + t] = (__bf16)cv[j][r];
    }
  }
}

// ---------------------------------------------------------------------------
// Flash attention: one wave handles one (b, h, 32-query tile); streams key
// blocks of 32 with online softmax.  K/V fragments are reused by two
// 16-row query tiles (16 WMMAs per 16 b128 key/value loads).
// ---------------------------------------------------------------------------
__global__ __launch_bounds__(32) void flash_attn(
    const __bf16* __restrict__ q, const __bf16* __restrict__ kmat,
    const __bf16* __restrict__ vt, __bf16* __restrict__ ao) {
  __shared__ __bf16 pl[32][40];          // P bounce buffer, padded rows

  const int qt   = blockIdx.x;           // 0..31 (T/32)
  const int h    = blockIdx.y;
  const int b    = blockIdx.z;
  const int lane = threadIdx.x & 31;
  const int n    = lane & 15;
  const int ro   = (lane >> 4) << 3;

  const __bf16* qb = q    + (size_t)(b * HH + h) * TT * DHH;
  const __bf16* kb = kmat + (size_t)(b * HH + h) * TT * DHH;
  const __bf16* vb = vt   + (size_t)(b * HH + h) * DHH * TT;

  v16bf qa[2][2];                        // [row tile][K chunk]
#pragma unroll
  for (int u = 0; u < 2; ++u) {
    qa[u][0] = load_frag_g(qb, DHH, qt * 32 + u * 16, 0);
    qa[u][1] = load_frag_g(qb, DHH, qt * 32 + u * 16, 32);
  }

  v8f o[2][4] = {};
  float mrow[2][8], lrow[2][8];
#pragma unroll
  for (int u = 0; u < 2; ++u)
#pragma unroll
    for (int r = 0; r < 8; ++r) { mrow[u][r] = -__builtin_inff(); lrow[u][r] = 0.f; }
  const float scale = 0.125f;            // 1/sqrt(DH)

  const int nkb = qt + 1;                // causal: key block start <= last row
  for (int kblk = 0; kblk < nkb; ++kblk) {
    const int key0 = kblk * 32;

    v16bf kf[2][2];                      // [key tile][K chunk]
#pragma unroll
    for (int w = 0; w < 2; ++w) {
      kf[w][0] = load_frag_g(kb, DHH, key0 + w * 16, 0);
      kf[w][1] = load_frag_g(kb, DHH, key0 + w * 16, 32);
    }

    v8f s[2][2] = {};                    // [row tile][key tile]
#pragma unroll
    for (int u = 0; u < 2; ++u)
#pragma unroll
      for (int w = 0; w < 2; ++w) {
        s[u][w] = wmma_bf16(qa[u][0], kf[w][0], s[u][w]);
        s[u][w] = wmma_bf16(qa[u][1], kf[w][1], s[u][w]);
      }

#pragma unroll
    for (int u = 0; u < 2; ++u) {
      float p0[8], p1[8], corr[8];
#pragma unroll
      for (int r = 0; r < 8; ++r) {
        const int rowg = qt * 32 + u * 16 + ro + r;
        const int c0   = key0 + n;
        const int c1   = c0 + 16;
        float a0 = (c0 <= rowg) ? s[u][0][r] * scale : -__builtin_inff();
        float a1 = (c1 <= rowg) ? s[u][1][r] * scale : -__builtin_inff();
        float bm = fmaxf(a0, a1);
#pragma unroll
        for (int off = 1; off < 16; off <<= 1)
          bm = fmaxf(bm, __shfl_xor(bm, off, 16));
        const float mnew = fmaxf(mrow[u][r], bm);
        const float cr   = __expf(mrow[u][r] - mnew);
        float e0 = __expf(a0 - mnew);
        float e1 = __expf(a1 - mnew);
        float bs = e0 + e1;
#pragma unroll
        for (int off = 1; off < 16; off <<= 1)
          bs += __shfl_xor(bs, off, 16);
        lrow[u][r] = lrow[u][r] * cr + bs;
        mrow[u][r] = mnew;
        corr[r]    = cr;
        p0[r] = e0;
        p1[r] = e1;
      }
#pragma unroll
      for (int r = 0; r < 8; ++r) {
        o[u][0][r] *= corr[r]; o[u][1][r] *= corr[r];
        o[u][2][r] *= corr[r]; o[u][3][r] *= corr[r];
      }
      // C-layout P -> row-major LDS tile
#pragma unroll
      for (int r = 0; r < 8; ++r) {
        pl[u * 16 + ro + r][n]      = (__bf16)p0[r];
        pl[u * 16 + ro + r][n + 16] = (__bf16)p1[r];
      }
    }
    __syncthreads();
    v16bf pa[2];
#pragma unroll
    for (int u = 0; u < 2; ++u)
      pa[u] = load_frag_lds(&pl[0][0], 40, u * 16);
    __syncthreads();

    // O += P (16x32) * V (32x64); B^T = V^T rows contiguous in vt
    v16bf vf[4];
#pragma unroll
    for (int j = 0; j < 4; ++j)
      vf[j] = load_frag_g(vb, TT, j * 16, key0);
#pragma unroll
    for (int u = 0; u < 2; ++u)
#pragma unroll
      for (int j = 0; j < 4; ++j)
        o[u][j] = wmma_bf16(pa[u], vf[j], o[u][j]);
  }

  // normalize and store concatenated-head layout ao[b, t, h*DH + d]
#pragma unroll
  for (int u = 0; u < 2; ++u)
#pragma unroll
    for (int r = 0; r < 8; ++r) {
      const float inv = 1.0f / lrow[u][r];
      const int t = qt * 32 + u * 16 + ro + r;
      __bf16* dst = ao + ((size_t)b * TT + t) * EE + h * DHH;
      dst[0  + n] = (__bf16)(o[u][0][r] * inv);
      dst[16 + n] = (__bf16)(o[u][1][r] * inv);
      dst[32 + n] = (__bf16)(o[u][2][r] * inv);
      dst[48 + n] = (__bf16)(o[u][3][r] * inv);
    }
}

// ---------------------------------------------------------------------------
// Output projection: out = ao (B*T x E) * Wp^T + bp, fp32 out.
// 4-wave workgroup computes a 64x64 tile.  The Wp K-slab is staged into a
// double-buffered LDS tile with GLOBAL_LOAD_ASYNC_TO_LDS_B128 and consumed
// by all 4 waves (B fetched from global once per workgroup).
// ---------------------------------------------------------------------------
__global__ __launch_bounds__(128) void out_proj(
    const __bf16* __restrict__ ao, const __bf16* __restrict__ wp,
    const float* __restrict__ bp, float* __restrict__ out) {
  __shared__ __bf16 bw[2][64][40];       // padded rows: conflict-free ds_load_b128

  const int tn    = blockIdx.x;          // 0..15  (E/64)
  const int tm    = blockIdx.y;          // 0..127 (B*T/64)
  const int tid   = threadIdx.x;         // 0..127
  const int w     = tid >> 5;            // wave id
  const int lane  = tid & 31;

  // staging assignment: 2 threads per Wp row, 32B each
  const int srow  = tid >> 1;            // 0..63
  const int shalf = tid & 1;             // 16-element half
  const __bf16* gB = wp + (size_t)(tn * 64 + srow) * EE + shalf * 16;

  const int arow0 = tm * 64 + w * 16;
  v8f acc[4] = {};

  // prologue: stage slab 0
  async_copy16(gB, &bw[0][srow][shalf * 16]);
  async_copy16(gB + 8, &bw[0][srow][shalf * 16 + 8]);

  const int NK = EE / 32;                // 32 K-steps
  for (int i = 0; i < NK; ++i) {
    const int k0 = i * 32;
    if (i + 1 < NK) {                    // prefetch next slab into other buffer
      const __bf16* gn = gB + k0 + 32;
      __bf16* ln = &bw[(i + 1) & 1][srow][shalf * 16];
      async_copy16(gn, ln);
      async_copy16(gn + 8, ln + 8);
      asm volatile("s_wait_asynccnt 0x2" ::: "memory");  // current slab done
    } else {
      asm volatile("s_wait_asynccnt 0x0" ::: "memory");
    }
    __syncthreads();

    v16bf a = load_frag_g(ao, EE, arow0, k0);
    const __bf16* lb = &bw[i & 1][0][0];
#pragma unroll
    for (int j = 0; j < 4; ++j) {
      v16bf bf = load_frag_lds(lb, 40, j * 16);
      acc[j] = wmma_bf16(a, bf, acc[j]);
    }
    __syncthreads();                     // all waves done before buffer reuse
  }

  const int n  = lane & 15;
  const int ro = (lane >> 4) << 3;
#pragma unroll
  for (int j = 0; j < 4; ++j) {
    const float bias = bp[tn * 64 + j * 16 + n];
#pragma unroll
    for (int r = 0; r < 8; ++r)
      out[(size_t)(arow0 + ro + r) * EE + tn * 64 + j * 16 + n] = acc[j][r] + bias;
  }
}

// ---------------------------------------------------------------------------
extern "C" void kernel_launch(void* const* d_in, const int* in_sizes, int n_in,
                              void* d_out, int out_size, void* d_ws,
                              size_t ws_size, hipStream_t stream) {
  (void)in_sizes; (void)n_in; (void)out_size; (void)ws_size;

  const float* x  = (const float*)d_in[0];
  const float* Wq = (const float*)d_in[1];
  const float* Wk = (const float*)d_in[2];
  const float* Wv = (const float*)d_in[3];
  const float* Wp = (const float*)d_in[4];
  const float* bp = (const float*)d_in[5];
  float* out = (float*)d_out;

  // workspace carve-up (all region sizes are multiples of 256B)
  char* ws = (char*)d_ws;
  size_t off = 0;
  const size_t szX  = (size_t)BB * TT * EE * sizeof(__bf16);         // 16 MB
  const size_t szW  = (size_t)HH * EE * DHH * sizeof(__bf16);        // 2 MB
  const size_t szWp = (size_t)EE * EE * sizeof(__bf16);              // 2 MB
  const size_t szH  = (size_t)BB * HH * TT * DHH * sizeof(__bf16);   // 16 MB

  __bf16* xb  = (__bf16*)(ws + off); off += szX;
  __bf16* wqt = (__bf16*)(ws + off); off += szW;
  __bf16* wkt = (__bf16*)(ws + off); off += szW;
  __bf16* wvt = (__bf16*)(ws + off); off += szW;
  __bf16* wpb = (__bf16*)(ws + off); off += szWp;
  __bf16* qb  = (__bf16*)(ws + off); off += szH;
  __bf16* kbf = (__bf16*)(ws + off); off += szH;
  __bf16* vtb = (__bf16*)(ws + off); off += szH;
  __bf16* aob = (__bf16*)(ws + off); off += szX;

  // 1) precision conversion + weight transposes
  cvt_f32_to_bf16<<<4096, 256, 0, stream>>>(x, xb, (long)BB * TT * EE);
  cvt_w_transpose<<<2048, 256, 0, stream>>>(Wq, wqt, (long)HH * EE * DHH);
  cvt_w_transpose<<<2048, 256, 0, stream>>>(Wk, wkt, (long)HH * EE * DHH);
  cvt_w_transpose<<<2048, 256, 0, stream>>>(Wv, wvt, (long)HH * EE * DHH);
  cvt_f32_to_bf16<<<2048, 256, 0, stream>>>(Wp, wpb, (long)EE * EE);

  // 2) fused QKV projections (V stored transposed)
  qkv_gemm_fused<<<dim3(TT / 16, HH, BB), 32, 0, stream>>>(
      xb, wqt, wkt, wvt, qb, kbf, vtb);

  // 3) fused causal flash attention (32 query rows per wave)
  flash_attn<<<dim3(TT / 32, HH, BB), 32, 0, stream>>>(qb, kbf, vtb, aob);

  // 4) output projection + bias (async-LDS staged weights)
  out_proj<<<dim3(EE / 64, (BB * TT) / 64), 128, 0, stream>>>(aob, wpb, bp, out);
}
